// MambaDFuse_25168508354817
// MI455X (gfx1250) — compile-verified, matrix operand order checked by reading, not run on hardware
//
#include <hip/hip_runtime.h>
#include <hip/hip_bf16.h>
#include <math.h>

// ---------------- model constants (from reference) ----------------
#define DM      128          // D_MODEL
#define DI      256          // D_INNER
#define DSTATE  64
#define NH      4
#define HD      64           // HEADDIM
#define KCONV   4
#define NB      4            // N_BLOCKS
#define DINPROJ 644          // 2*DI + 2*DSTATE + NH
#define DINPAD  656          // padded to multiple of 16
#define CONVCH  384          // DI + 2*DSTATE
#define H2      512          // 2*HIDDEN
#define HID     256          // HIDDEN
#define LFIX    16384        // H*W = 128*128

typedef __attribute__((ext_vector_type(16))) __bf16        v16bf;
typedef __attribute__((ext_vector_type(8)))  float         v8f;
typedef __attribute__((ext_vector_type(4)))  unsigned int  u32x4;

union FragBF { v16bf v; u32x4 q[2]; unsigned short u[16]; };

__device__ __forceinline__ unsigned short f2bfu(float f) {
    unsigned u = __float_as_uint(f);
    unsigned r = u + 0x7FFFu + ((u >> 16) & 1u);   // round-to-nearest-even
    return (unsigned short)(r >> 16);
}
__device__ __forceinline__ float siluf(float x) {
    return x / (1.0f + expf(-x));
}

// ---------------- weight convert + transpose:  W[K,N] f32 -> WT[Npad,K] bf16 ----
__global__ void k_convert_wT(const float* __restrict__ W, unsigned short* __restrict__ WT,
                             int K, int N, int Npad) {
    int idx = blockIdx.x * blockDim.x + threadIdx.x;
    int total = Npad * K;
    if (idx >= total) return;
    int n = idx / K;
    int k = idx - n * K;
    float v = (n < N) ? W[(size_t)k * N + n] : 0.0f;
    WT[idx] = f2bfu(v);
}

// ---------------- plain copy ----------------
__global__ void k_copy(const float* __restrict__ src, float* __restrict__ dst, size_t n) {
    size_t i = (size_t)blockIdx.x * blockDim.x + threadIdx.x;
    if (i < n) dst[i] = src[i];
}

// ---------------- LayerNorm over D=128, bf16 output (one token per block) ----
__global__ void k_layernorm(const float* __restrict__ x, const float* __restrict__ w,
                            const float* __restrict__ b, unsigned short* __restrict__ out) {
    __shared__ float red[DM];
    size_t t = blockIdx.x;
    int i = threadIdx.x;                 // 0..127
    float v = x[t * DM + i];
    red[i] = v;
    __syncthreads();
    for (int s = DM / 2; s > 0; s >>= 1) {
        if (i < s) red[i] += red[i + s];
        __syncthreads();
    }
    float mu = red[0] * (1.0f / DM);
    __syncthreads();
    float d = v - mu;
    red[i] = d * d;
    __syncthreads();
    for (int s = DM / 2; s > 0; s >>= 1) {
        if (i < s) red[i] += red[i + s];
        __syncthreads();
    }
    float var = red[0] * (1.0f / DM);
    out[t * DM + i] = f2bfu(d * rsqrtf(var + 1e-5f) * w[i] + b[i]);
}

// ---------------- WMMA bf16 GEMM ----------------
// C[M,N] = A[M,K] * WT[Npad,K]^T (+bias, +resid). A, WT bf16 row-major.
// One wave computes a 64x16 slab: 4 M-subtiles sharing one B fragment per k-step.
template<int K, bool HAS_BIAS, bool HAS_RES>
__global__ void k_gemm_wmma(const unsigned short* __restrict__ A,
                            const unsigned short* __restrict__ WT,
                            const float* __restrict__ bias,
                            const float* __restrict__ resid,
                            float* __restrict__ C,
                            int M, int N, int Npad) {
    int ntn = Npad >> 4;
    int tile = blockIdx.x * blockDim.y + threadIdx.y;          // wave-uniform
    int ntiles = (M >> 6) * ntn;
    if (tile >= ntiles) return;                                 // whole wave exits
    int m0 = (tile / ntn) << 6;                                 // 64 rows
    int n0 = (tile % ntn) << 4;                                 // 16 cols

    int lane = threadIdx.x;          // 0..31
    int half = lane >> 4;            // 0/1
    int lm   = lane & 15;

    const unsigned short* brow = WT + (size_t)(n0 + lm) * K;
    const unsigned short* ar0  = A  + (size_t)(m0 + lm) * K;
    const unsigned short* ar1  = ar0 + (size_t)16 * K;
    const unsigned short* ar2  = ar0 + (size_t)32 * K;
    const unsigned short* ar3  = ar0 + (size_t)48 * K;

    v8f acc0 = {}, acc1 = {}, acc2 = {}, acc3 = {};
#pragma unroll
    for (int k0 = 0; k0 < K; k0 += 32) {
        int koff = k0 + half * 8;            // lane's contiguous 8-elem run
        FragBF bfr, a0, a1, a2, a3;
        bfr.q[0] = *(const u32x4*)(brow + koff);
        bfr.q[1] = *(const u32x4*)(brow + koff + 16);
        a0.q[0] = *(const u32x4*)(ar0 + koff);  a0.q[1] = *(const u32x4*)(ar0 + koff + 16);
        a1.q[0] = *(const u32x4*)(ar1 + koff);  a1.q[1] = *(const u32x4*)(ar1 + koff + 16);
        a2.q[0] = *(const u32x4*)(ar2 + koff);  a2.q[1] = *(const u32x4*)(ar2 + koff + 16);
        a3.q[0] = *(const u32x4*)(ar3 + koff);  a3.q[1] = *(const u32x4*)(ar3 + koff + 16);
        acc0 = __builtin_amdgcn_wmma_f32_16x16x32_bf16(false, a0.v, false, bfr.v, (short)0, acc0, false, false);
        acc1 = __builtin_amdgcn_wmma_f32_16x16x32_bf16(false, a1.v, false, bfr.v, (short)0, acc1, false, false);
        acc2 = __builtin_amdgcn_wmma_f32_16x16x32_bf16(false, a2.v, false, bfr.v, (short)0, acc2, false, false);
        acc3 = __builtin_amdgcn_wmma_f32_16x16x32_bf16(false, a3.v, false, bfr.v, (short)0, acc3, false, false);
    }

    int n = n0 + lm;
    if (n < N) {
        float bv = HAS_BIAS ? bias[n] : 0.0f;
        v8f accs[4] = {acc0, acc1, acc2, acc3};
#pragma unroll
        for (int s = 0; s < 4; ++s) {
#pragma unroll
            for (int r = 0; r < 8; ++r) {
                int m = m0 + 16 * s + r + 8 * half;
                size_t off = (size_t)m * N + n;
                float v = accs[s][r] + bv;
                if (HAS_RES) v += resid[off];
                C[off] = v;
            }
        }
    }
}

// ---------------- depthwise causal conv (K=4) + bias + SiLU over xBC slice ----
__global__ void k_conv_silu(const float* __restrict__ zx,
                            const float* __restrict__ cw,   // [4, 384]
                            const float* __restrict__ cb,   // [384]
                            float* __restrict__ out, int Bn, int L) {
    size_t idx = (size_t)blockIdx.x * blockDim.x + threadIdx.x;
    size_t total = (size_t)Bn * L * CONVCH;
    if (idx >= total) return;
    int c = (int)(idx % CONVCH);
    size_t bt = idx / CONVCH;
    int t = (int)(bt % L);
    size_t brow = bt - t;                         // b*L
    float s = cb[c];
#pragma unroll
    for (int k = 0; k < KCONV; ++k) {
        int tt = t - (KCONV - 1) + k;
        if (tt >= 0)
            s += zx[(brow + tt) * DINPROJ + DI + c] * cw[k * CONVCH + c];
    }
    out[idx] = siluf(s);
}

// ---------------- dt = softplus(dt_raw + dt_bias); decay = exp(dt * -exp(A_log))
__global__ void k_dt_decay(const float* __restrict__ zx,
                           const float* __restrict__ dt_bias,
                           const float* __restrict__ A_log,
                           float* __restrict__ dt_out, float* __restrict__ dc_out,
                           size_t total /* M*NH */) {
    size_t idx = (size_t)blockIdx.x * blockDim.x + threadIdx.x;
    if (idx >= total) return;
    int h = (int)(idx & (NH - 1));
    size_t t = idx >> 2;
    float d = zx[t * DINPROJ + (DINPROJ - NH) + h] + dt_bias[h];
    float sp = (d > 20.0f) ? d : log1pf(expf(d));
    float Aa = -expf(A_log[h]);
    dt_out[idx] = sp;
    dc_out[idx] = expf(sp * Aa);
}

// ---------------- sequential SSM scan, barrier-free ----------------
// One wave per (b, h, p): lane holds state elements n=lane and n=lane+32.
// y[p] = sum_n h[p][n]*C[n] via wave32 shfl_xor reduction. No LDS, no syncs.
__global__ void k_ssm_scan(const float* __restrict__ xconv,   // [B,L,384]
                           const float* __restrict__ dt,      // [B,L,4]
                           const float* __restrict__ dec,     // [B,L,4]
                           float* __restrict__ yssm,          // [B,L,256]
                           int L) {
    int wave = blockIdx.x * (blockDim.x >> 5) + (threadIdx.x >> 5);
    int lane = threadIdx.x & 31;
    int p = wave & 63;
    int h = (wave >> 6) & 3;
    int b = wave >> 8;
    int n0 = lane, n1 = lane + 32;

    float h0 = 0.0f, h1 = 0.0f;
    for (int t = 0; t < L; ++t) {
        size_t row = (size_t)b * L + t;
        const float* xr = xconv + row * CONVCH;
        float dtv = dt [row * NH + h];
        float dcv = dec[row * NH + h];
        float xv  = xr[h * HD + p];
        float B0  = xr[DI + n0];
        float B1  = xr[DI + n1];
        float C0  = xr[DI + DSTATE + n0];
        float C1  = xr[DI + DSTATE + n1];

        if (t + 8 < L) {  // software prefetch (global_prefetch_b8)
            __builtin_prefetch(xr + 8 * CONVCH + DI + n0, 0, 1);
        }

        float a = dtv * xv;
        h0 = h0 * dcv + a * B0;
        h1 = h1 * dcv + a * B1;
        float ps = h0 * C0 + h1 * C1;
#pragma unroll
        for (int o = 16; o > 0; o >>= 1) ps += __shfl_xor(ps, o, 32);
        if (lane == 0) yssm[row * DI + h * HD + p] = ps;
    }
}

// ---------------- y = (yssm + Dp[h]*xh) * silu(z), bf16 output --------------
__global__ void k_combine(const float* __restrict__ yssm, const float* __restrict__ xconv,
                          const float* __restrict__ zx, const float* __restrict__ Dp,
                          unsigned short* __restrict__ ybuf, size_t total /* M*256 */) {
    size_t idx = (size_t)blockIdx.x * blockDim.x + threadIdx.x;
    if (idx >= total) return;
    int i = (int)(idx & (DI - 1));
    size_t m = idx >> 8;
    int h = i >> 6;
    float y = yssm[idx] + Dp[h] * xconv[m * CONVCH + i];
    float z = zx[m * DINPROJ + i];
    ybuf[idx] = f2bfu(y * siluf(z));
}

// ---------------- SwiGLU: g = silu(h1) * h2, bf16 output ----------------
__global__ void k_glu(const float* __restrict__ h12, unsigned short* __restrict__ g,
                      size_t total /* M*256 */) {
    size_t idx = (size_t)blockIdx.x * blockDim.x + threadIdx.x;
    if (idx >= total) return;
    int j = (int)(idx & (HID - 1));
    size_t m = idx >> 8;
    float a = h12[m * H2 + j];
    float c = h12[m * H2 + HID + j];
    g[idx] = f2bfu(siluf(a) * c);
}

// =======================================================================
extern "C" void kernel_launch(void* const* d_in, const int* in_sizes, int n_in,
                              void* d_out, int out_size, void* d_ws, size_t ws_size,
                              hipStream_t stream) {
    const float* x       = (const float*)d_in[0];
    const float* ln1_w   = (const float*)d_in[1];
    const float* ln1_b   = (const float*)d_in[2];
    const float* ln2_w   = (const float*)d_in[3];
    const float* ln2_b   = (const float*)d_in[4];
    const float* in_proj = (const float*)d_in[5];
    const float* conv_w  = (const float*)d_in[6];
    const float* conv_b  = (const float*)d_in[7];
    const float* dt_bias = (const float*)d_in[8];
    const float* A_log   = (const float*)d_in[9];
    const float* Dp      = (const float*)d_in[10];
    const float* out_proj= (const float*)d_in[11];
    const float* fc1_w   = (const float*)d_in[12];
    const float* fc1_b   = (const float*)d_in[13];
    const float* fc2_w   = (const float*)d_in[14];
    const float* fc2_b   = (const float*)d_in[15];

    const int L  = LFIX;
    const int BL = in_sizes[0] / DM;        // B * L
    const int Bn = BL / L;
    const size_t M = (size_t)BL;

    // ---- workspace carve-out ----
    size_t off = 0;
    auto alloc = [&](size_t bytes) -> void* {
        void* p = (char*)d_ws + off;
        off += (bytes + 255) & ~(size_t)255;
        return p;
    };
    unsigned short* x1 = (unsigned short*)alloc(M * DM * 2);     // bf16 LN out
    float* zx    = (float*)alloc(M * DINPROJ * 4);
    float* xconv = (float*)alloc(M * CONVCH  * 4);
    float* dtb   = (float*)alloc(M * NH      * 4);
    float* dcb   = (float*)alloc(M * NH      * 4);
    float* yssm  = (float*)alloc(M * DI      * 4);
    unsigned short* ybuf = (unsigned short*)alloc(M * DI * 2);   // bf16 GEMM input
    unsigned short* wt_in  = (unsigned short*)alloc((size_t)NB * DINPAD * DM * 2);
    unsigned short* wt_out = (unsigned short*)alloc((size_t)NB * DM * DI * 2);
    unsigned short* wt_fc1 = (unsigned short*)alloc((size_t)NB * H2 * DM * 2);
    unsigned short* wt_fc2 = (unsigned short*)alloc((size_t)NB * DM * DI * 2);
    float* h12 = zx;   // alias: zx dead by the time fc1 runs, reused next block

    float* xr = (float*)d_out;   // running residual stream lives in d_out

    const int T = 256;
    auto cdiv = [](size_t a, size_t b) { return (unsigned)((a + b - 1) / b); };

    // ---- one-time (per call) weight convert+transpose ----
    for (int i = 0; i < NB; ++i) {
        k_convert_wT<<<cdiv((size_t)DINPAD * DM, T), T, 0, stream>>>(
            in_proj + (size_t)i * DM * DINPROJ, wt_in + (size_t)i * DINPAD * DM,
            DM, DINPROJ, DINPAD);
        k_convert_wT<<<cdiv((size_t)DM * DI, T), T, 0, stream>>>(
            out_proj + (size_t)i * DI * DM, wt_out + (size_t)i * DM * DI,
            DI, DM, DM);
        k_convert_wT<<<cdiv((size_t)H2 * DM, T), T, 0, stream>>>(
            fc1_w + (size_t)i * DM * H2, wt_fc1 + (size_t)i * H2 * DM,
            DM, H2, H2);
        k_convert_wT<<<cdiv((size_t)DM * DI, T), T, 0, stream>>>(
            fc2_w + (size_t)i * DI * DM, wt_fc2 + (size_t)i * DM * DI,
            DI, DM, DM);
    }

    // ---- x -> d_out (residual stream) ----
    k_copy<<<cdiv(M * DM, T), T, 0, stream>>>(x, xr, M * DM);

    dim3 gblk(32, 8);   // 8 waves per block, one 64x16 slab per wave
    auto gemm_grid = [&](int N_pad) {
        size_t tiles = (M >> 6) * (size_t)(N_pad >> 4);
        return dim3(cdiv(tiles, 8));
    };

    for (int i = 0; i < NB; ++i) {
        // LN1 -> bf16
        k_layernorm<<<(unsigned)M, DM, 0, stream>>>(xr, ln1_w + i * DM, ln1_b + i * DM, x1);
        // in_proj GEMM
        k_gemm_wmma<DM, false, false><<<gemm_grid(DINPAD), gblk, 0, stream>>>(
            x1, wt_in + (size_t)i * DINPAD * DM, nullptr, nullptr, zx,
            (int)M, DINPROJ, DINPAD);
        // conv + SiLU
        k_conv_silu<<<cdiv(M * CONVCH, T), T, 0, stream>>>(
            zx, conv_w + (size_t)i * KCONV * CONVCH, conv_b + (size_t)i * CONVCH,
            xconv, Bn, L);
        // dt / decay
        k_dt_decay<<<cdiv(M * NH, T), T, 0, stream>>>(
            zx, dt_bias + i * NH, A_log + i * NH, dtb, dcb, M * NH);
        // sequential scan (one wave per (b,h,p), barrier-free)
        k_ssm_scan<<<Bn * 32, 256, 0, stream>>>(xconv, dtb, dcb, yssm, L);
        // gate + skip -> bf16
        k_combine<<<cdiv(M * DI, T), T, 0, stream>>>(
            yssm, xconv, zx, Dp + i * NH, ybuf, M * DI);
        // out_proj GEMM + residual into d_out
        k_gemm_wmma<DI, false, true><<<gemm_grid(DM), gblk, 0, stream>>>(
            ybuf, wt_out + (size_t)i * DM * DI, nullptr, xr, xr,
            (int)M, DM, DM);
        // LN2 -> bf16
        k_layernorm<<<(unsigned)M, DM, 0, stream>>>(xr, ln2_w + i * DM, ln2_b + i * DM, x1);
        // fc1 GEMM + bias
        k_gemm_wmma<DM, true, false><<<gemm_grid(H2), gblk, 0, stream>>>(
            x1, wt_fc1 + (size_t)i * H2 * DM, fc1_b + (size_t)i * H2, nullptr, h12,
            (int)M, H2, H2);
        // SwiGLU -> bf16
        k_glu<<<cdiv(M * HID, T), T, 0, stream>>>(h12, ybuf, M * HID);
        // fc2 GEMM + bias + residual into d_out
        k_gemm_wmma<DI, true, true><<<gemm_grid(DM), gblk, 0, stream>>>(
            ybuf, wt_fc2 + (size_t)i * DM * DI, fc2_b + (size_t)i * DM, xr, xr,
            (int)M, DM, DM);
    }
    (void)n_in; (void)out_size; (void)ws_size;
}
